// IntelligentGateNetwork_23957327577722
// MI455X (gfx1250) — matrix operand order repaired and checked by reading, not compiled, not run on hardware
//
#include <hip/hip_runtime.h>

typedef __attribute__((ext_vector_type(16))) _Float16 v16h;
typedef __attribute__((ext_vector_type(8)))  _Float16 v8h;
typedef __attribute__((ext_vector_type(8)))  float    v8f;

#define BATCH 8192
#define TT 64
#define VV 64
#define DD 128
#define HH 32
#define G4 128   // 4*H

// Branch-free activations on the native v_exp_f32 / v_rcp_f32 path.
__device__ __forceinline__ float frcp(float x) { return __builtin_amdgcn_rcpf(x); }
__device__ __forceinline__ float sigf(float x) { return frcp(1.0f + __expf(-x)); }
__device__ __forceinline__ float tanhfast(float x) {
    // tanh(x) = (e^{2x}-1)/(e^{2x}+1) = 1 - 2/(1+e^{2x});  saturates correctly at +/-inf
    return 1.0f - 2.0f * frcp(1.0f + __expf(2.0f * x));
}

// ---------------------------------------------------------------------------
// Kernel 1: proj[v][j] = sum_d emb[v,d] * w_ih[j,d] + b[j]   (V=64 x 4H=128)
// ---------------------------------------------------------------------------
__global__ void proj_kernel(const float* __restrict__ emb,
                            const float* __restrict__ wih,
                            const float* __restrict__ bias,
                            float* __restrict__ proj) {
    int idx = blockIdx.x * 256 + threadIdx.x;   // 0..8191
    int v = idx >> 7;
    int j = idx & 127;
    const float* e = emb + v * DD;
    const float* w = wih + j * DD;
    float s = bias[j];
    #pragma unroll 8
    for (int d = 0; d < DD; ++d) s += e[d] * w[d];
    proj[v * G4 + j] = s;
}

// ---------------------------------------------------------------------------
// Kernel 2: LSTM scan.  One wave (32 lanes) = one 16-row batch tile, one dir.
// g(16x128) = h(16x32) @ w_hh.T  via 8x v_wmma_f32_16x16x32_f16, with the
// C accumulator pre-loaded with the gathered input projection xp.
// ---------------------------------------------------------------------------
__global__ __launch_bounds__(32, 1)
void lstm_kernel(const int*   __restrict__ x,
                 const float* __restrict__ projF, const float* __restrict__ projB,
                 const float* __restrict__ whhF,  const float* __restrict__ whhB,
                 float* __restrict__ houtF,       float* __restrict__ houtB) {
    const int tile = blockIdx.x >> 1;
    const int dir  = blockIdx.x & 1;
    const float* proj = dir ? projB : projF;
    const float* whh  = dir ? whhB  : whhF;
    float*       hout = dir ? houtB : houtF;

    const int lane = threadIdx.x;
    const int half = lane >> 4;     // 0: lanes 0-15, 1: lanes 16-31
    const int lm   = lane & 15;

    __shared__ int ldsX[16 * TT];                    // token tile
    __shared__ __align__(16) _Float16 ldsH[16 * HH]; // h state, row-major (m,k)

    // Stage the x tile (16 rows x 64 steps) into LDS once.
    for (int i = lane; i < 16 * TT; i += 32) {
        int m = i >> 6, t = i & 63;
        ldsX[i] = x[(tile * 16 + m) * TT + t];
    }

    // Resident B operands: B[k][n] = w_hh[n_global][k], tile nt covers n=nt*16..+15.
    // Lane layout (f16 32x16 B): lanes 0-15 -> K=0..15 col n=lm; lanes 16-31 -> K=16..31.
    v16h bw[8];
    #pragma unroll
    for (int nt = 0; nt < 8; ++nt) {
        const float* wr = whh + (nt * 16 + lm) * HH + half * 16;
        #pragma unroll
        for (int e = 0; e < 16; ++e) bw[nt][e] = (_Float16)wr[e];
    }

    for (int i = lane; i < 16 * HH; i += 32) ldsH[i] = (_Float16)0.0f;
    v8f cst[2] = {};            // cell state, tiles j=0,1 (D-layout)
    float hl[2][8] = {};
    __syncthreads();

    for (int t = 0; t < TT; ++t) {
        const int tidx = dir ? (TT - 1 - t) : t;

        // Tokens for this lane's 8 M rows (C layout: VGPR r -> M = r + 8*half).
        int tok[8];
        #pragma unroll
        for (int r = 0; r < 8; ++r) tok[r] = ldsX[(8 * half + r) * TT + tidx];

        // A operand (h, 16x32 f16): lane holds row m=lm,
        // K = half*16 + 0..15 (elems 0-15), matching B.
        v16h a;
        {
            const v8h lo = *(const v8h*)&ldsH[lm * HH + half * 16];
            const v8h hi = *(const v8h*)&ldsH[lm * HH + half * 16 + 8];
            #pragma unroll
            for (int e = 0; e < 8; ++e) { a[e] = lo[e]; a[8 + e] = hi[e]; }
        }

        // Gather xp into C and do the recurrent GEMM: g = xp + h @ W^T.
        v8f g[8];
        #pragma unroll
        for (int nt = 0; nt < 8; ++nt) {
            v8f cinit;
            #pragma unroll
            for (int r = 0; r < 8; ++r)
                cinit[r] = proj[tok[r] * G4 + nt * 16 + lm];
            g[nt] = __builtin_amdgcn_wmma_f32_16x16x32_f16(
                false, a, false, bw[nt], (short)0, cinit, false, false);
        }

        __syncthreads();   // all lanes finished reading ldsH before overwrite

        // Gates: i=tiles 0,1; f=2,3; g~=4,5; o=6,7.  (N-tile j spans hidden cols j*16..+15)
        #pragma unroll
        for (int j = 0; j < 2; ++j) {
            #pragma unroll
            for (int r = 0; r < 8; ++r) {
                float ig = sigf(g[0 + j][r]);
                float fg = sigf(g[2 + j][r]);
                float gg = tanhfast(g[4 + j][r]);
                float og = sigf(g[6 + j][r]);
                float cn = fg * cst[j][r] + ig * gg;
                cst[j][r] = cn;
                float hn = og * tanhfast(cn);
                hl[j][r] = hn;
                // row-major (m,k): m = r + 8*half, k = j*16 + lm
                ldsH[(8 * half + r) * HH + j * 16 + lm] = (_Float16)hn;
            }
        }
        __syncthreads();
    }

    // Final hidden state -> global (f32).
    #pragma unroll
    for (int j = 0; j < 2; ++j)
        #pragma unroll
        for (int r = 0; r < 8; ++r)
            hout[(tile * 16 + 8 * half + r) * HH + j * 16 + lm] = hl[j][r];
}

// ---------------------------------------------------------------------------
// Kernel 3: features + gate MLP, one thread per batch row.
// ---------------------------------------------------------------------------
__global__ __launch_bounds__(32)
void feat_kernel(const int* __restrict__ x,
                 const float* __restrict__ hF, const float* __restrict__ hB,
                 const float* __restrict__ ent_w, const float* __restrict__ ent_b,
                 const float* __restrict__ len_w, const float* __restrict__ len_b,
                 const float* __restrict__ cd_w,  const float* __restrict__ cd_b,
                 const float* __restrict__ pat_w, const float* __restrict__ pat_b,
                 const float* __restrict__ g1_w,  const float* __restrict__ g1_b,
                 const float* __restrict__ g2_w,  const float* __restrict__ g2_b,
                 const float* __restrict__ g3_w,  const float* __restrict__ g3_b,
                 float* __restrict__ out) {
    const int tid = threadIdx.x;
    const int row = blockIdx.x * 32 + tid;

    __shared__ float sH[32 * 65];    // per-thread histogram/probs (stride 65: conflict-free)
    __shared__ float sF[32 * 105];   // per-thread feature vector (stride 105)
    __shared__ float s1[32 * 65];    // per-thread hidden layer 1 (stride 65)
    float* hst = &sH[tid * 65];
    float* ft  = &sF[tid * 105];
    float* h1  = &s1[tid * 65];

    for (int v = 0; v < VV; ++v) hst[v] = 0.0f;

    // Histogram + compacted-sequence pattern stats in one pass.
    int L = 0, prev1 = -1, prev2 = -1;
    int rep = 0, inc = 0, dec = 0, per = 0;
    for (int t = 0; t < TT; ++t) {
        int tk = x[row * TT + t];
        if (tk != 0) {
            hst[tk] += 1.0f;
            if (L >= 1) { rep += (tk == prev1); inc += (tk > prev1); dec += (tk < prev1); }
            if (L >= 2) { per += (tk == prev2); }
            prev2 = prev1; prev1 = tk; ++L;
        }
    }

    const float Lf = (float)L;
    const float denom = fmaxf(Lf, 1.0f);
    float ent = 0.0f;
    for (int v = 0; v < VV; ++v) {
        float p = hst[v] / denom;
        hst[v] = p;
        ent -= p * __logf(p + 1e-8f);
    }
    if (L <= 1) ent = 0.0f;

    // feat = [h_fwd(32), h_bwd(32), ent(8), len(8), cd(16), pat(8)]
    for (int k = 0; k < HH; ++k) ft[k]      = hF[row * HH + k];
    for (int k = 0; k < HH; ++k) ft[32 + k] = hB[row * HH + k];
    const float e4  = ent * 0.25f;
    for (int j = 0; j < 8; ++j)  ft[64 + j] = e4 * ent_w[j] + ent_b[j];
    const float l40 = Lf / 40.0f;
    for (int j = 0; j < 8; ++j)  ft[72 + j] = l40 * len_w[j] + len_b[j];
    for (int j = 0; j < 16; ++j) {
        float s = cd_b[j];
        for (int v = 0; v < VV; ++v) s += hst[v] * cd_w[j * VV + v];
        ft[80 + j] = s;
    }
    const float lm1 = fmaxf(Lf - 1.0f, 1.0f);
    const float lm2 = fmaxf(Lf - 2.0f, 1.0f);
    float pat[4];
    pat[0] = (float)rep / lm1;
    pat[1] = (float)inc / lm1;
    pat[2] = (float)dec / lm1;
    pat[3] = (L >= 4) ? (float)per / lm2 : 0.0f;
    if (L <= 1) { pat[0] = pat[1] = pat[2] = pat[3] = 0.0f; }
    for (int j = 0; j < 8; ++j) {
        float s = pat_b[j];
        #pragma unroll
        for (int k = 0; k < 4; ++k) s += pat[k] * pat_w[j * 4 + k];
        ft[96 + j] = s;
    }

    // MLP 104 -> 64 -> 32 -> 2, softmax.
    for (int j = 0; j < 64; ++j) {
        float s = g1_b[j];
        for (int k = 0; k < 104; ++k) s += ft[k] * g1_w[j * 104 + k];
        h1[j] = fmaxf(s, 0.0f);
    }
    float z0 = g3_b[0], z1 = g3_b[1];
    for (int j = 0; j < 32; ++j) {
        float s = g2_b[j];
        for (int k = 0; k < 64; ++k) s += h1[k] * g2_w[j * 64 + k];
        float r = fmaxf(s, 0.0f);
        z0 += r * g3_w[j];
        z1 += r * g3_w[32 + j];
    }
    const float m  = fmaxf(z0, z1);
    const float e0 = __expf(z0 - m);
    const float e1 = __expf(z1 - m);
    const float inv = frcp(e0 + e1);
    out[row * 2 + 0] = e0 * inv;
    out[row * 2 + 1] = e1 * inv;
}

// ---------------------------------------------------------------------------
extern "C" void kernel_launch(void* const* d_in, const int* in_sizes, int n_in,
                              void* d_out, int out_size, void* d_ws, size_t ws_size,
                              hipStream_t stream) {
    const int*   x      = (const int*)  d_in[0];
    const float* emb    = (const float*)d_in[1];
    const float* w_ih_f = (const float*)d_in[2];
    const float* w_hh_f = (const float*)d_in[3];
    const float* b_f    = (const float*)d_in[4];
    const float* w_ih_b = (const float*)d_in[5];
    const float* w_hh_b = (const float*)d_in[6];
    const float* b_b    = (const float*)d_in[7];
    const float* ent_w  = (const float*)d_in[8];
    const float* ent_b  = (const float*)d_in[9];
    const float* len_w  = (const float*)d_in[10];
    const float* len_b  = (const float*)d_in[11];
    const float* cd_w   = (const float*)d_in[12];
    const float* cd_b   = (const float*)d_in[13];
    const float* pat_w  = (const float*)d_in[14];
    const float* pat_b  = (const float*)d_in[15];
    const float* g1_w   = (const float*)d_in[16];
    const float* g1_b   = (const float*)d_in[17];
    const float* g2_w   = (const float*)d_in[18];
    const float* g2_b   = (const float*)d_in[19];
    const float* g3_w   = (const float*)d_in[20];
    const float* g3_b   = (const float*)d_in[21];
    float* out = (float*)d_out;

    float* projF = (float*)d_ws;                  // 64*128
    float* projB = projF + VV * G4;               // 64*128
    float* hF    = projB + VV * G4;               // 8192*32
    float* hB    = hF + BATCH * HH;               // 8192*32

    proj_kernel<<<32, 256, 0, stream>>>(emb, w_ih_f, b_f, projF);
    proj_kernel<<<32, 256, 0, stream>>>(emb, w_ih_b, b_b, projB);
    lstm_kernel<<<(BATCH / 16) * 2, 32, 0, stream>>>(x, projF, projB, w_hh_f, w_hh_b, hF, hB);
    feat_kernel<<<BATCH / 32, 32, 0, stream>>>(x, hF, hB, ent_w, ent_b, len_w, len_b,
                                               cd_w, cd_b, pat_w, pat_b,
                                               g1_w, g1_b, g2_w, g2_b, g3_w, g3_b, out);
}
